// Item2ItemNet_661424964277
// MI455X (gfx1250) — matrix operand order repaired and checked by reading, not compiled, not run on hardware
//
#include <hip/hip_runtime.h>
#include <hip/hip_bf16.h>
#include <stdint.h>

typedef __attribute__((ext_vector_type(16))) _Float16 v16h;
typedef __attribute__((ext_vector_type(8)))  _Float16 v8h;
typedef __attribute__((ext_vector_type(4)))  _Float16 v4h;
typedef __attribute__((ext_vector_type(8)))  float    v8f;

#define B_N 16384
#define S_N 50
#define D_N 64

// ---------------- small prep kernels (fp32, negligible cost) ----------------

// q0[b][d] = target_emb[b] @ Wq[0:64, d]
__global__ void prep_q(const float* __restrict__ target, const float* __restrict__ Wq,
                       float* __restrict__ q0) {
  __shared__ float t[64];
  const int b = blockIdx.x, d = threadIdx.x;
  t[d] = target[(size_t)b * 64 + d];
  __syncthreads();
  float acc = 0.f;
  #pragma unroll 8
  for (int c = 0; c < 64; ++c) acc += t[c] * Wq[c * 64 + d];
  q0[(size_t)b * 64 + d] = acc;
}

// pcorr[s][d] = bq[d] + pos_emb[s] @ Wq[64:68, d]
__global__ void prep_pos(const float* __restrict__ pos, const float* __restrict__ Wq,
                         const float* __restrict__ bq, float* __restrict__ pc) {
  const int s = blockIdx.x, d = threadIdx.x;
  float acc = bq[d];
  #pragma unroll
  for (int j = 0; j < 4; ++j) acc += pos[s * 4 + j] * Wq[(64 + j) * 64 + d];
  pc[s * 64 + d] = acc;
}

// Repack effective W1 (192x80: [q | k | q*k] coefficients) into WMMA B-fragment
// linear layout: block (ks,nt) -> 32 lanes x 16 f16. Lane = khalf*16 + n%16,
// element e -> K = ks*32 + khalf*16 + e.
__global__ void prep_w1(const float* __restrict__ W1, _Float16* __restrict__ out) {
  const int idx = blockIdx.x * 512 + threadIdx.x;   // 30 blocks * 512 = 15360
  const int blk = idx >> 9, rem = idx & 511;
  const int lane = rem >> 4, e = rem & 15;
  const int ks = blk / 5, nt = blk % 5;
  const int k = ks * 32 + ((lane >> 4) << 4) + e;   // 0..191
  const int n = nt * 16 + (lane & 15);              // 0..79 (pad beyond)
  float v = 0.f;
  if (n < 80) {
    if (k < 64)        v = W1[k * 80 + n] + W1[(128 + k) * 80 + n];     // q coeff
    else if (k < 128)  v = W1[k * 80 + n] - W1[(k + 64) * 80 + n];      // k coeff
    else               v = W1[(k + 64) * 80 + n];                       // q*k coeff
  }
  out[idx] = (_Float16)v;
}

// W2 (80x40) padded to K=96, N=48, same B-fragment layout.
__global__ void prep_w2(const float* __restrict__ W2, _Float16* __restrict__ out) {
  const int idx = blockIdx.x * 512 + threadIdx.x;   // 9 blocks * 512 = 4608
  const int blk = idx >> 9, rem = idx & 511;
  const int lane = rem >> 4, e = rem & 15;
  const int ks = blk / 3, nt = blk % 3;
  const int k = ks * 32 + ((lane >> 4) << 4) + e;
  const int n = nt * 16 + (lane & 15);
  float v = 0.f;
  if (k < 80 && n < 40) v = W2[k * 40 + n];
  out[idx] = (_Float16)v;
}

// ---------------- fused main kernel ----------------

__device__ __forceinline__ float sigmoidf_(float x) {
  return 1.0f / (1.0f + __expf(-x));
}

__device__ __forceinline__ v16h cat16(v8h lo, v8h hi) {
  return __builtin_shufflevector(lo, hi, 0,1,2,3,4,5,6,7,8,9,10,11,12,13,14,15);
}

#define WB1_ELEMS (6 * 5 * 512)   // 15360 f16
#define WB2_ELEMS (3 * 3 * 512)   // 4608 f16
#define PER_WAVE_BYTES 50176
#define SMEM_BYTES (39936 + 4 * PER_WAVE_BYTES)   // 240640

__global__ __launch_bounds__(128, 1)
void i2i_fused(const float* __restrict__ seq, const int* __restrict__ mask,
               const float* __restrict__ q0g, const float* __restrict__ pcorr,
               const float* __restrict__ alpha,
               const _Float16* __restrict__ w1b, const _Float16* __restrict__ w2b,
               const float* __restrict__ b1g, const float* __restrict__ b2g,
               const float* __restrict__ W3g, const float* __restrict__ b3g,
               float* __restrict__ attn, float* __restrict__ rel) {
  extern __shared__ char smem[];
  _Float16* WB1 = (_Float16*)smem;
  _Float16* WB2 = WB1 + WB1_ELEMS;

  const int tid  = threadIdx.x;
  const int wave = tid >> 5;
  const int lane = tid & 31;

  // cooperative copy of the shared WMMA B-fragment tables into LDS
  {
    const uint32_t* s1 = (const uint32_t*)w1b;
    uint32_t*       d1 = (uint32_t*)WB1;
    for (int i = tid; i < WB1_ELEMS / 2; i += 128) d1[i] = s1[i];
    const uint32_t* s2 = (const uint32_t*)w2b;
    uint32_t*       d2 = (uint32_t*)WB2;
    for (int i = tid; i < WB2_ELEMS / 2; i += 128) d2[i] = s2[i];
  }
  __syncthreads();

  const int b = blockIdx.x * 4 + wave;
  char* wbase = smem + 39936 + wave * PER_WAVE_BYTES;
  _Float16* Abuf = (_Float16*)wbase;          // [64][192] f16: q | k | q*k
  _Float16* Hbuf = Abuf + 64 * 192;           // [64][96]  f16: sigmoid(layer1), K-padded
  float*    Sbuf = (float*)(Hbuf + 64 * 96);  // [50][64]  fp32 sequence_emb copy
  float*    scoreBuf = Sbuf + S_N * 64;       // [64]
  float*    pBuf     = scoreBuf + 64;         // [64]

  // ---- build phase: Abuf (f16), Sbuf (fp32), zero Hbuf pad columns ----
  const float4* q0row = (const float4*)(q0g + (size_t)b * 64);
  const float4* al4   = (const float4*)alpha;
  #pragma unroll
  for (int rr = 0; rr < 2; ++rr) {
    const int s = lane + rr * 32;             // row 0..63 (rows >= 50 are zero pad)
    if (s < S_N) {
      const float4* sp = (const float4*)(seq + ((size_t)b * S_N + s) * 64);
      const float4* pp = (const float4*)(pcorr + s * 64);
      for (int c = 0; c < 16; ++c) {
        float4 kv = sp[c];
        float4 pv = pp[c];
        float4 qz = q0row[c];
        float4 av = al4[c];
        float qx = qz.x + pv.x; qx = (qx >= 0.f) ? qx : av.x * qx;
        float qy = qz.y + pv.y; qy = (qy >= 0.f) ? qy : av.y * qy;
        float qw2 = qz.z + pv.z; qw2 = (qw2 >= 0.f) ? qw2 : av.z * qw2;
        float qv2 = qz.w + pv.w; qv2 = (qv2 >= 0.f) ? qv2 : av.w * qv2;
        _Float16* arow = Abuf + s * 192 + c * 4;
        v4h tq = { (_Float16)qx, (_Float16)qy, (_Float16)qw2, (_Float16)qv2 };
        v4h tk = { (_Float16)kv.x, (_Float16)kv.y, (_Float16)kv.z, (_Float16)kv.w };
        v4h tm = { (_Float16)(qx * kv.x), (_Float16)(qy * kv.y),
                   (_Float16)(qw2 * kv.z), (_Float16)(qv2 * kv.w) };
        *(v4h*)(arow)       = tq;
        *(v4h*)(arow + 64)  = tk;
        *(v4h*)(arow + 128) = tm;
        ((float4*)(Sbuf + s * 64))[c] = kv;
      }
    } else {
      v4h z = { (_Float16)0.f, (_Float16)0.f, (_Float16)0.f, (_Float16)0.f };
      for (int c = 0; c < 16; ++c) {
        _Float16* arow = Abuf + s * 192 + c * 4;
        *(v4h*)(arow) = z; *(v4h*)(arow + 64) = z; *(v4h*)(arow + 128) = z;
      }
    }
    // zero Hbuf pad columns 80..95 (K padding for layer 2)
    float4 z4 = { 0.f, 0.f, 0.f, 0.f };
    *(float4*)(Hbuf + s * 96 + 80) = z4;
    *(float4*)(Hbuf + s * 96 + 88) = z4;
  }
  __syncthreads();

  const int hi8 = (lane >> 4) << 3;           // 0 for lanes 0-15, 8 for lanes 16-31

  // ---- layer 1: [64 x 192] @ [192 x 80] via v_wmma_f32_16x16x32_f16 ----
  float b1v[5];
  #pragma unroll
  for (int nt = 0; nt < 5; ++nt) {
    const int n = nt * 16 + (lane & 15);
    b1v[nt] = (n < 80) ? b1g[n] : 0.f;
  }
  for (int mt = 0; mt < 4; ++mt) {
    v8f C[5];
    #pragma unroll
    for (int nt = 0; nt < 5; ++nt)
      #pragma unroll
      for (int r = 0; r < 8; ++r) C[nt][r] = b1v[nt];
    #pragma unroll
    for (int ks = 0; ks < 6; ++ks) {
      const _Float16* ap = Abuf + (mt * 16 + (lane & 15)) * 192 + ks * 32 + (hi8 << 1);
      v16h A = cat16(*(const v8h*)ap, *(const v8h*)(ap + 8));
      #pragma unroll
      for (int nt = 0; nt < 5; ++nt) {
        const _Float16* bp = WB1 + ((ks * 5 + nt) * 32 + lane) * 16;
        v16h Bf = cat16(*(const v8h*)bp, *(const v8h*)(bp + 8));
        C[nt] = __builtin_amdgcn_wmma_f32_16x16x32_f16(
            false, A, false, Bf, (short)0, C[nt], false, false);
      }
    }
    // sigmoid + scatter into Hbuf (D-layout -> row-major for layer-2 A)
    const int rowb = mt * 16 + hi8;
    #pragma unroll
    for (int nt = 0; nt < 5; ++nt) {
      const int col = nt * 16 + (lane & 15);
      #pragma unroll
      for (int r = 0; r < 8; ++r)
        Hbuf[(rowb + r) * 96 + col] = (_Float16)sigmoidf_(C[nt][r]);
    }
  }
  __syncthreads();

  // ---- layer 2 + layer 3: [64 x 96] @ [96 x 48], then in-register 40-dot ----
  float b2v[3], w3v[3];
  #pragma unroll
  for (int nt = 0; nt < 3; ++nt) {
    const int n = nt * 16 + (lane & 15);
    b2v[nt] = (n < 40) ? b2g[n] : 0.f;
    w3v[nt] = (n < 40) ? W3g[n] : 0.f;
  }
  const float b3v = b3g[0];
  for (int mt = 0; mt < 4; ++mt) {
    v8f C2[3];
    #pragma unroll
    for (int nt = 0; nt < 3; ++nt)
      #pragma unroll
      for (int r = 0; r < 8; ++r) C2[nt][r] = b2v[nt];
    #pragma unroll
    for (int ks = 0; ks < 3; ++ks) {
      const _Float16* ap = Hbuf + (mt * 16 + (lane & 15)) * 96 + ks * 32 + (hi8 << 1);
      v16h A = cat16(*(const v8h*)ap, *(const v8h*)(ap + 8));
      #pragma unroll
      for (int nt = 0; nt < 3; ++nt) {
        const _Float16* bp = WB2 + ((ks * 3 + nt) * 32 + lane) * 16;
        v16h Bf = cat16(*(const v8h*)bp, *(const v8h*)(bp + 8));
        C2[nt] = __builtin_amdgcn_wmma_f32_16x16x32_f16(
            false, A, false, Bf, (short)0, C2[nt], false, false);
      }
    }
    #pragma unroll
    for (int r = 0; r < 8; ++r) {
      float acc = sigmoidf_(C2[0][r]) * w3v[0]
                + sigmoidf_(C2[1][r]) * w3v[1]
                + sigmoidf_(C2[2][r]) * w3v[2];
      acc += __shfl_xor(acc, 1, 32);
      acc += __shfl_xor(acc, 2, 32);
      acc += __shfl_xor(acc, 4, 32);
      acc += __shfl_xor(acc, 8, 32);          // sum within each 16-lane half (N dim)
      if ((lane & 15) == 0) scoreBuf[mt * 16 + hi8 + r] = acc + b3v;
    }
  }
  __syncthreads();

  // ---- masked softmax, rel_i2i, weighted sum over sequence_emb ----
  const int s0 = lane;
  const int s1 = lane + 32;
  const bool v1 = (s1 < S_N);
  const int m0 = mask[(size_t)b * S_N + s0];
  const int m1 = v1 ? mask[(size_t)b * S_N + s1] : 0;
  const float sc0 = scoreBuf[s0];
  const float sc1 = v1 ? scoreBuf[s1] : 0.f;
  const float a0 = (m0 != 0) ? sc0 : -1e9f;
  const float a1 = (m1 != 0) ? sc1 : -1e9f;

  float mx = v1 ? fmaxf(a0, a1) : a0;
  #pragma unroll
  for (int d = 1; d < 32; d <<= 1) mx = fmaxf(mx, __shfl_xor(mx, d, 32));
  const float e0 = __expf(a0 - mx);
  const float e1 = v1 ? __expf(a1 - mx) : 0.f;
  float sm = e0 + e1;
  #pragma unroll
  for (int d = 1; d < 32; d <<= 1) sm += __shfl_xor(sm, d, 32);
  const float inv = 1.0f / sm;
  pBuf[s0] = e0 * inv;
  if (v1) pBuf[s1] = e1 * inv;

  float rl = ((m0 != 0) ? sc0 : 0.f) + ((v1 && m1 != 0) ? sc1 : 0.f);
  #pragma unroll
  for (int d = 1; d < 32; d <<= 1) rl += __shfl_xor(rl, d, 32);
  __syncthreads();

  float acc0 = 0.f, acc1 = 0.f;
  for (int s = 0; s < S_N; ++s) {
    const float ps = pBuf[s];
    acc0 += ps * Sbuf[s * 64 + lane];
    acc1 += ps * Sbuf[s * 64 + lane + 32];
  }
  attn[(size_t)b * 64 + lane]      = acc0;
  attn[(size_t)b * 64 + lane + 32] = acc1;
  if (lane == 0) rel[b] = rl;
}

// ---------------- launch ----------------

extern "C" void kernel_launch(void* const* d_in, const int* in_sizes, int n_in,
                              void* d_out, int out_size, void* d_ws, size_t ws_size,
                              hipStream_t stream) {
  const float* target = (const float*)d_in[0];
  const float* seq    = (const float*)d_in[1];
  const int*   mask   = (const int*)d_in[2];
  const float* pos    = (const float*)d_in[3];
  const float* Wq     = (const float*)d_in[4];
  const float* bq     = (const float*)d_in[5];
  const float* alpha  = (const float*)d_in[6];
  const float* W1     = (const float*)d_in[7];
  const float* b1     = (const float*)d_in[8];
  const float* W2     = (const float*)d_in[9];
  const float* b2     = (const float*)d_in[10];
  const float* W3     = (const float*)d_in[11];
  const float* b3     = (const float*)d_in[12];

  float*    q0  = (float*)d_ws;                       // B*64 f32
  float*    pc  = q0 + (size_t)B_N * 64;              // S*64 f32
  _Float16* w1b = (_Float16*)(pc + S_N * 64);         // 15360 f16
  _Float16* w2b = w1b + WB1_ELEMS;                    // 4608 f16

  float* attn = (float*)d_out;
  float* relv = attn + (size_t)B_N * 64;

  prep_q  <<<B_N, 64, 0, stream>>>(target, Wq, q0);
  prep_pos<<<S_N, 64, 0, stream>>>(pos, Wq, bq, pc);
  prep_w1 <<<30, 512, 0, stream>>>(W1, w1b);
  prep_w2 <<<9, 512, 0, stream>>>(W2, w2b);

  i2i_fused<<<B_N / 4, 128, SMEM_BYTES, stream>>>(
      seq, mask, q0, pc, alpha, w1b, w2b, b1, b2, W3, b3, attn, relv);
}
